// MultiHeadedAttention_RPR_35424890258016
// MI455X (gfx1250) — matrix operand order repaired and checked by reading, not compiled, not run on hardware
//
#include <hip/hip_runtime.h>
#include <hip/hip_fp16.h>

// ---------------------------------------------------------------------------
// MultiHeadedAttention with relative position representations (Shaw et al.)
// MI455X / gfx1250: wave32, v_wmma_f32_16x16x32_f16, async global->LDS staging
// with double buffering, LDS float atomics for the relative-V binning.
// ---------------------------------------------------------------------------

typedef __attribute__((ext_vector_type(16))) _Float16 v16h;
typedef __attribute__((ext_vector_type(4)))  _Float16 v4h;
typedef __attribute__((ext_vector_type(8)))  float    v8f;
typedef __attribute__((ext_vector_type(4)))  int      v4i;

#define WMMA_F32_F16(a, b, c) \
  __builtin_amdgcn_wmma_f32_16x16x32_f16(false, (a), false, (b), (short)0, (c), false, false)

#if defined(__gfx1250__) && __has_builtin(__builtin_amdgcn_global_load_async_to_lds_b128) && \
    __has_builtin(__builtin_amdgcn_s_wait_asynccnt)
#define HAVE_ASYNC_LDS 1
#endif

// Copy 16 bytes global -> LDS. Async (ASYNCcnt-tracked) when available.
__device__ __forceinline__ void copy_b128_g2l(const void* g, void* l) {
#ifdef HAVE_ASYNC_LDS
  __builtin_amdgcn_global_load_async_to_lds_b128(
      (__attribute__((address_space(1))) v4i*)(uintptr_t)g,
      (__attribute__((address_space(3))) v4i*)(uintptr_t)l, 0, 0);
#else
  *(uint4*)l = *(const uint4*)g;
#endif
}
__device__ __forceinline__ void async_wait0() {
#ifdef HAVE_ASYNC_LDS
  __builtin_amdgcn_s_wait_asynccnt(0);
#endif
}

constexpr int cB   = 4;
constexpr int cS   = 1024;
constexpr int cH   = 1024;
constexpr int cNH  = 16;
constexpr int cDK  = 64;            // cH / cNH
constexpr int cMRP = 64;
constexpr int cVOC = 2 * cMRP + 1;  // 129
constexpr int cVP  = 132;           // padded vocab stride
constexpr int cBH  = cB * cNH;      // 64
constexpr int cNT  = cS / 32;       // 32 key tiles

// A-fragment element -> K index (16-bit A 16x32, contiguous k = 16*hi + e)
__device__ __forceinline__ int a_frag_k(int e, int hi) {
  return (hi << 4) + e;
}

// ---------------------------------------------------------------------------
// Kernel 1: projection GEMM  dst(f16,[BH][S][DK]) = x(4096x1024) @ W^T + bias
// block = 128 threads (4 waves); block tile 64x64; wave tile 16x64.
// ---------------------------------------------------------------------------
__global__ __launch_bounds__(128) void qkv_proj_kernel(
    const float* __restrict__ x, const float* __restrict__ W,
    const float* __restrict__ bias, _Float16* __restrict__ dst) {
  __shared__ alignas(16) _Float16 sX[64 * 32];
  __shared__ alignas(16) _Float16 sW[64 * 32];
  const int tid  = threadIdx.x;
  const int lane = tid & 31, wave = tid >> 5;
  const int m = lane & 15, hi = lane >> 4;
  const int row0 = blockIdx.x * 64;   // over B*S = 4096
  const int col0 = blockIdx.y * 64;   // over H = 1024

  v8f acc[4] = {};

  for (int k0 = 0; k0 < cH; k0 += 32) {
    // stage 64x32 f32 tiles of x and W as f16 (float4 loads, packed stores)
    for (int i = tid; i < 512; i += 128) {
      int r = i >> 3, kc = (i & 7) << 2;
      float4 xv = *(const float4*)&x[(size_t)(row0 + r) * cH + k0 + kc];
      float4 wv = *(const float4*)&W[(size_t)(col0 + r) * cH + k0 + kc];
      v4h xh = {(_Float16)xv.x, (_Float16)xv.y, (_Float16)xv.z, (_Float16)xv.w};
      v4h wh = {(_Float16)wv.x, (_Float16)wv.y, (_Float16)wv.z, (_Float16)wv.w};
      *(v4h*)&sX[r * 32 + kc] = xh;
      *(v4h*)&sW[r * 32 + kc] = wh;
    }
    __syncthreads();
    if (k0 + 32 < cH)
      __builtin_prefetch(&x[(size_t)(row0 + wave * 16) * cH + k0 + 32], 0, 0);

    v16h a;
#pragma unroll
    for (int e = 0; e < 16; ++e)
      a[e] = sX[(wave * 16 + m) * 32 + a_frag_k(e, hi)];

#pragma unroll
    for (int nc = 0; nc < 4; ++nc) {
      v16h bf;
#pragma unroll
      for (int e = 0; e < 16; ++e)
        bf[e] = sW[(nc * 16 + m) * 32 + hi * 16 + e];  // B[k,n] = W[n, k0+k]
      acc[nc] = WMMA_F32_F16(a, bf, acc[nc]);
    }
    __syncthreads();
  }

  // store to head-major f16 layout [BH][S][DK]
#pragma unroll
  for (int nc = 0; nc < 4; ++nc) {
    int gc = col0 + nc * 16 + m;  // output feature
    int h = gc >> 6, dk = gc & 63;
    float bv = bias[gc];
#pragma unroll
    for (int j = 0; j < 8; ++j) {
      int gr = row0 + wave * 16 + j + hi * 8;  // over B*S
      int bb = gr >> 10, ss = gr & 1023;
      dst[(((size_t)(bb * cNH + h)) * cS + ss) * cDK + dk] = (_Float16)(acc[nc][j] + bv);
    }
  }
}

// ---------------------------------------------------------------------------
// Kernel 2: s_rel[bh,q,v] = Q[bh,q,:] . embK[v,:]   (V=129, DK=64)
// ---------------------------------------------------------------------------
__global__ __launch_bounds__(160) void srel_kernel(
    const _Float16* __restrict__ Qh, const float* __restrict__ embK,
    float* __restrict__ srel) {
  __shared__ float sq[cDK];
  const int bq = blockIdx.x;  // bh*S + q
  const int v  = threadIdx.x;
  if (v < cDK) sq[v] = (float)Qh[(size_t)bq * cDK + v];
  __syncthreads();
  if (v < cVOC) {
    float acc = 0.f;
#pragma unroll 8
    for (int d = 0; d < cDK; ++d) acc += sq[d] * embK[v * cDK + d];
    srel[(size_t)bq * cVP + v] = acc;
  }
}

// ---------------------------------------------------------------------------
// Kernel 3: fused attention. 1 block = 4 waves = 64 query rows of one (b,h).
// Wave owns 16 q rows. Two-pass softmax streaming over 32-key tiles with
// async double-buffered K/V staging:
//   pass A: row maxima of  S = Q K^T + s_rel gather
//   pass B: P = exp(S - M); l += P; wsum[bin] += P (ds_add_f32);
//           O += P V (WMMA);  then O = (O + wsum @ embV) / l
// ---------------------------------------------------------------------------
__global__ __launch_bounds__(128) void attn_kernel(
    const _Float16* __restrict__ Qh, const _Float16* __restrict__ Kh,
    const _Float16* __restrict__ Vh, const float* __restrict__ srel,
    const float* __restrict__ embV, _Float16* __restrict__ attn) {
  __shared__ alignas(16) _Float16 sK[2][32 * cDK];   // 2 x 4 KB
  __shared__ alignas(16) _Float16 sV[2][32 * cDK];   // 2 x 4 KB
  __shared__ alignas(16) _Float16 sP[4][16 * 32];    // 4 KB
  __shared__ float sWsum[4][16 * cVP];               // 33 KB

  const int tid  = threadIdx.x;
  const int lane = tid & 31, wave = tid >> 5;
  const int m = lane & 15, hi = lane >> 4;
  const int bh = blockIdx.x >> 4;  // / (S/64)
  const int q0 = (blockIdx.x & 15) * 64 + wave * 16;
  const int b  = bh / cNH, h = bh % cNH;

  for (int i = lane; i < 16 * cVP; i += 32) sWsum[wave][i] = 0.f;

  const _Float16* Kbh = Kh + (size_t)bh * cS * cDK;
  const _Float16* Vbh = Vh + (size_t)bh * cS * cDK;
  const float*    srb = srel + (size_t)bh * cS * cVP;

  // tile stagers: 32 keys x 64 dk f16 = 4 KB = 256 x b128
  auto stageK = [&](int kt, int buf) {
    const char* src = (const char*)(Kbh + (size_t)kt * 32 * cDK);
    char* dst = (char*)sK[buf];
    for (int i = tid; i < 256; i += 128) copy_b128_g2l(src + i * 16, dst + i * 16);
  };
  auto stageV = [&](int kt, int buf) {
    const char* src = (const char*)(Vbh + (size_t)kt * 32 * cDK);
    char* dst = (char*)sV[buf];
    for (int i = tid; i < 256; i += 128) copy_b128_g2l(src + i * 16, dst + i * 16);
  };

  // Q A-fragments (two 16x32 K-chunks covering DK=64), resident in VGPRs.
  v16h aq[2];
  {
    const _Float16* qb = Qh + ((size_t)bh * cS + q0) * cDK;
#pragma unroll
    for (int c = 0; c < 2; ++c)
#pragma unroll
      for (int e = 0; e < 16; ++e)
        aq[c][e] = qb[m * cDK + c * 32 + a_frag_k(e, hi)];
  }

  float mx[8], lsum[8];
#pragma unroll
  for (int j = 0; j < 8; ++j) { mx[j] = -3.0e38f; lsum[j] = 0.f; }
  v8f o[4] = {};

  // ---------------- pass A: row maxima ----------------
  stageK(0, 0);
  for (int kt = 0; kt < cNT; ++kt) {
    async_wait0();
    __syncthreads();
    if (kt + 1 < cNT) stageK(kt + 1, (kt + 1) & 1);  // overlap next fetch
    if (kt + 2 < cNT) __builtin_prefetch(Kbh + (size_t)(kt + 2) * 32 * cDK, 0, 0);
    const _Float16* K_ = sK[kt & 1];

#pragma unroll
    for (int ko = 0; ko < 32; ko += 16) {
      v8f acc = {};
#pragma unroll
      for (int c = 0; c < 2; ++c) {
        v16h bK;
#pragma unroll
        for (int e = 0; e < 16; ++e)
          bK[e] = K_[(ko + m) * cDK + c * 32 + hi * 16 + e];  // B[k,key]=K[key,dk]
        acc = WMMA_F32_F16(aq[c], bK, acc);
      }
      int kg = kt * 32 + ko + m;
#pragma unroll
      for (int j = 0; j < 8; ++j) {
        int qg = q0 + j + hi * 8;
        int dist = kg - qg;
        dist = dist < -cMRP ? -cMRP : (dist > cMRP ? cMRP : dist);
        float s = acc[j] + srb[(size_t)qg * cVP + dist + cMRP];
        mx[j] = fmaxf(mx[j], s);
      }
    }
  }
  // rows live across 16-lane half-waves: reduce within width-16 groups
#pragma unroll
  for (int j = 0; j < 8; ++j) {
    float t = mx[j];
    for (int off = 8; off > 0; off >>= 1) t = fmaxf(t, __shfl_xor(t, off, 16));
    mx[j] = t;
  }

  // ---------------- pass B: P, l, wsum, O += P V ----------------
  stageK(0, 0);
  stageV(0, 0);
  for (int kt = 0; kt < cNT; ++kt) {
    async_wait0();
    __syncthreads();
    if (kt + 1 < cNT) { stageK(kt + 1, (kt + 1) & 1); stageV(kt + 1, (kt + 1) & 1); }
    const _Float16* K_ = sK[kt & 1];
    const _Float16* V_ = sV[kt & 1];

#pragma unroll
    for (int ko = 0; ko < 32; ko += 16) {
      v8f acc = {};
#pragma unroll
      for (int c = 0; c < 2; ++c) {
        v16h bK;
#pragma unroll
        for (int e = 0; e < 16; ++e)
          bK[e] = K_[(ko + m) * cDK + c * 32 + hi * 16 + e];
        acc = WMMA_F32_F16(aq[c], bK, acc);
      }
      int kg = kt * 32 + ko + m;
#pragma unroll
      for (int j = 0; j < 8; ++j) {
        int mrow = j + hi * 8;
        int qg = q0 + mrow;
        int dist = kg - qg;
        dist = dist < -cMRP ? -cMRP : (dist > cMRP ? cMRP : dist);
        float s = acc[j] + srb[(size_t)qg * cVP + dist + cMRP];
        float p = __expf(s - mx[j]);
        lsum[j] += p;
        atomicAdd(&sWsum[wave][mrow * cVP + dist + cMRP], p);  // ds_add_f32
        sP[wave][mrow * 32 + ko + m] = (_Float16)p;
      }
    }
    __syncthreads();  // sP stores -> fragment loads (and uniform control flow)

    // O(16x64) += P(16x32) @ V(32x64)
    v16h aP;
#pragma unroll
    for (int e = 0; e < 16; ++e)
      aP[e] = sP[wave][m * 32 + a_frag_k(e, hi)];
#pragma unroll
    for (int dc = 0; dc < 4; ++dc) {
      v16h bV;
#pragma unroll
      for (int e = 0; e < 16; ++e)
        bV[e] = V_[(hi * 16 + e) * cDK + dc * 16 + m];
      o[dc] = WMMA_F32_F16(aP, bV, o[dc]);
    }
  }

  // reduce row sums across 16-lane groups
#pragma unroll
  for (int j = 0; j < 8; ++j) {
    float t = lsum[j];
    for (int off = 8; off > 0; off >>= 1) t += __shfl_xor(t, off, 16);
    lsum[j] = t;
  }
  __syncthreads();

  // relative-V term: rel[dc][j] = sum_v wsum[row][v] * embV[v][dc*16+n]
  float rel[4][8];
#pragma unroll
  for (int dc = 0; dc < 4; ++dc)
#pragma unroll
    for (int j = 0; j < 8; ++j) rel[dc][j] = 0.f;
  for (int v = 0; v < cVOC; ++v) {
    float ev[4];
#pragma unroll
    for (int dc = 0; dc < 4; ++dc) ev[dc] = embV[v * cDK + dc * 16 + m];
#pragma unroll
    for (int j = 0; j < 8; ++j) {
      float w = sWsum[wave][(j + hi * 8) * cVP + v];
#pragma unroll
      for (int dc = 0; dc < 4; ++dc) rel[dc][j] += w * ev[dc];
    }
  }

  // normalize and write concat-head f16 output attn[b][q][h*64+d]
#pragma unroll
  for (int dc = 0; dc < 4; ++dc) {
    int d = dc * 16 + m;
#pragma unroll
    for (int j = 0; j < 8; ++j) {
      int qg = q0 + j + hi * 8;
      float inv = 1.0f / lsum[j];
      attn[((size_t)b * cS + qg) * cH + h * cDK + d] =
          (_Float16)((o[dc][j] + rel[dc][j]) * inv);
    }
  }
}

// ---------------------------------------------------------------------------
// Kernel 4: output projection  out(f32) = attn(f16, 4096x1024) @ Wo^T + bo
// A tile staged with async b128 copies (already f16); W converted in VALU.
// ---------------------------------------------------------------------------
__global__ __launch_bounds__(128) void out_proj_kernel(
    const _Float16* __restrict__ A, const float* __restrict__ W,
    const float* __restrict__ bias, float* __restrict__ out) {
  __shared__ alignas(16) _Float16 sX[64 * 32];
  __shared__ alignas(16) _Float16 sW[64 * 32];
  const int tid  = threadIdx.x;
  const int lane = tid & 31, wave = tid >> 5;
  const int m = lane & 15, hi = lane >> 4;
  const int row0 = blockIdx.x * 64;
  const int col0 = blockIdx.y * 64;

  v8f acc[4] = {};

  for (int k0 = 0; k0 < cH; k0 += 32) {
    // async copy f16 A tile: 64 rows x 32 k = 4 KB = 256 x b128
    for (int i = tid; i < 256; i += 128) {
      int r = i >> 2, kc = (i & 3) << 3;
      copy_b128_g2l(&A[(size_t)(row0 + r) * cH + k0 + kc], &sX[r * 32 + kc]);
    }
    // convert f32 W tile
    for (int i = tid; i < 512; i += 128) {
      int r = i >> 3, kc = (i & 7) << 2;
      float4 wv = *(const float4*)&W[(size_t)(col0 + r) * cH + k0 + kc];
      v4h wh = {(_Float16)wv.x, (_Float16)wv.y, (_Float16)wv.z, (_Float16)wv.w};
      *(v4h*)&sW[r * 32 + kc] = wh;
    }
    async_wait0();
    __syncthreads();

    v16h a;
#pragma unroll
    for (int e = 0; e < 16; ++e)
      a[e] = sX[(wave * 16 + m) * 32 + a_frag_k(e, hi)];

#pragma unroll
    for (int nc = 0; nc < 4; ++nc) {
      v16h bf;
#pragma unroll
      for (int e = 0; e < 16; ++e)
        bf[e] = sW[(nc * 16 + m) * 32 + hi * 16 + e];
      acc[nc] = WMMA_F32_F16(a, bf, acc[nc]);
    }
    __syncthreads();
  }

#pragma unroll
  for (int nc = 0; nc < 4; ++nc) {
    int gc = col0 + nc * 16 + m;
    float bv = bias[gc];
#pragma unroll
    for (int j = 0; j < 8; ++j) {
      int gr = row0 + wave * 16 + j + hi * 8;
      out[(size_t)gr * cH + gc] = acc[nc][j] + bv;
    }
  }
}

// ---------------------------------------------------------------------------
extern "C" void kernel_launch(void* const* d_in, const int* in_sizes, int n_in,
                              void* d_out, int out_size, void* d_ws, size_t ws_size,
                              hipStream_t stream) {
  (void)in_sizes; (void)n_in; (void)out_size; (void)ws_size;
  const float* x    = (const float*)d_in[0];
  // d_in[1] = mask (dead in reference)
  const float* Wq   = (const float*)d_in[2];
  const float* bq   = (const float*)d_in[3];
  const float* Wk   = (const float*)d_in[4];
  const float* bk   = (const float*)d_in[5];
  const float* Wv   = (const float*)d_in[6];
  const float* bv   = (const float*)d_in[7];
  const float* Wo   = (const float*)d_in[8];
  const float* bo   = (const float*)d_in[9];
  const float* embK = (const float*)d_in[10];
  const float* embV = (const float*)d_in[11];
  float* out = (float*)d_out;

  char* ws = (char*)d_ws;
  const size_t qkvBytes  = (size_t)cBH * cS * cDK * sizeof(_Float16);  // 8 MB
  const size_t srelBytes = (size_t)cBH * cS * cVP * sizeof(float);     // 33 MB
  _Float16* Qh   = (_Float16*)(ws);
  _Float16* Kh   = (_Float16*)(ws + qkvBytes);
  _Float16* Vh   = (_Float16*)(ws + 2 * qkvBytes);
  float*    srel = (float*)(ws + 3 * qkvBytes);
  _Float16* attn = (_Float16*)(ws + 3 * qkvBytes + srelBytes);

  dim3 gemmGrid(cB * cS / 64, cH / 64);  // 64 x 16
  qkv_proj_kernel<<<gemmGrid, 128, 0, stream>>>(x, Wq, bq, Qh);
  qkv_proj_kernel<<<gemmGrid, 128, 0, stream>>>(x, Wk, bk, Kh);
  qkv_proj_kernel<<<gemmGrid, 128, 0, stream>>>(x, Wv, bv, Vh);
  srel_kernel<<<cBH * cS, 160, 0, stream>>>(Qh, embK, srel);
  attn_kernel<<<cBH * (cS / 64), 128, 0, stream>>>(Qh, Kh, Vh, srel, embV, attn);
  out_proj_kernel<<<gemmGrid, 128, 0, stream>>>(attn, Wo, bo, out);
}
